// BertAttention_41549513622120
// MI455X (gfx1250) — compile-verified, matrix operand order checked by reading, not run on hardware
//
#include <hip/hip_runtime.h>
#include <hip/hip_bf16.h>
#include <math.h>

#define DD   128
#define HH   8
#define DHH  16

typedef __attribute__((ext_vector_type(16))) __bf16 v16bf;
typedef __attribute__((ext_vector_type(8)))  __bf16 v8bf;
typedef __attribute__((ext_vector_type(8)))  float  v8f;

union Frag16 { v16bf f; v8bf h[2]; };

// ---------------------------------------------------------------------------
// GEMM: Out[n,128] = X[n,128] @ W[128,128] + bias (+ residual), bf16 WMMA.
// One block = 16 output rows x 128 cols; 8 waves, each wave owns a 16x16 tile.
// K-loop: 4 x V_WMMA_F32_16X16X32_BF16. W staged transposed in LDS so both
// A and B fragments are two contiguous 16-byte runs -> ds_load_b128 pairs.
// ---------------------------------------------------------------------------
template <bool HAS_RES>
__global__ __launch_bounds__(256)
void gemm_bf16_wmma(const float* __restrict__ X,
                    const float* __restrict__ W,
                    const float* __restrict__ bias,
                    const float* __restrict__ residual,
                    float* __restrict__ Out, int nRows) {
    __shared__ __bf16 sA[16 * DD];    //  4 KB : activation tile, row-major [m][k]
    __shared__ __bf16 sWt[DD * DD];   // 32 KB : weights transposed   [n][k]
    const int tid     = threadIdx.x;
    const int rowBase = blockIdx.x * 16;

    for (int i = tid; i < 16 * DD; i += 256) {
        int r = rowBase + (i >> 7);
        sA[i] = (__bf16)((r < nRows) ? X[r * DD + (i & 127)] : 0.0f);
    }
    for (int i = tid; i < DD * DD; i += 256) {
        int k = i >> 7, c = i & 127;            // W is [k][c] row-major in global
        sWt[c * DD + k] = (__bf16)W[i];
    }
    __syncthreads();

    const int lane = tid & 31;
    const int wave = tid >> 5;
    const int mrow = lane & 15;          // A row / B col within tile
    const int kb   = (lane >> 4) * 8;    // K sub-block select per lane half

    v8f acc = {};
#pragma unroll
    for (int s = 0; s < 4; ++s) {
        const int k0 = s * 32;
        // Per ISA 7.12.2: lane's 16 elements = K in [kb,kb+8) ++ [kb+16,kb+24)
        Frag16 a, b;
        a.h[0] = *(const v8bf*)(sA  + mrow * DD + k0 + kb);
        a.h[1] = *(const v8bf*)(sA  + mrow * DD + k0 + kb + 16);
        b.h[0] = *(const v8bf*)(sWt + (wave * 16 + mrow) * DD + k0 + kb);
        b.h[1] = *(const v8bf*)(sWt + (wave * 16 + mrow) * DD + k0 + kb + 16);
        acc = __builtin_amdgcn_wmma_f32_16x16x32_bf16(
                  false, a.f, false, b.f, (short)0, acc, false, false);
    }

    const int   ncol = wave * 16 + mrow;
    const int   mb   = (lane >> 4) * 8;
    const float bval = bias[ncol];

    if (rowBase + 16 <= nRows) {            // block-uniform fast path (always here)
#pragma unroll
        for (int r = 0; r < 8; ++r) {
            int   row = rowBase + mb + r;
            float val = acc[r] + bval;
            if (HAS_RES) val += residual[row * DD + ncol];
            Out[row * DD + ncol] = val;
        }
    } else {                                // tail path (defensive; unused for n%16==0)
#pragma unroll
        for (int r = 0; r < 8; ++r) {
            int row = rowBase + mb + r;
            if (row < nRows) {
                float val = acc[r] + bval;
                if (HAS_RES) val += residual[row * DD + ncol];
                Out[row * DD + ncol] = val;
            }
        }
    }
}

// ---------------------------------------------------------------------------
// Monotone float <-> uint keys so atomicMax(uint) implements float max.
// ---------------------------------------------------------------------------
__device__ __forceinline__ unsigned enc_key(float f) {
    unsigned b = __float_as_uint(f);
    return ((int)b < 0) ? ~b : (b | 0x80000000u);
}
__device__ __forceinline__ float dec_key(unsigned k) {
    unsigned b = (k & 0x80000000u) ? (k ^ 0x80000000u) : ~k;
    return __uint_as_float(b);
}

__global__ void init_scratch(float* __restrict__ agg, float* __restrict__ denom,
                             unsigned* __restrict__ mbuf, int n) {
    long i = (long)blockIdx.x * 256 + threadIdx.x;
    if (i < (long)n * DD) agg[i] = 0.0f;
    if (i < (long)n * HH) { denom[i] = 0.0f; mbuf[i] = 0u; }  // key 0 == -NaN (identity)
}

// scores[e,h] = dot16(K[src[e],h,:], Q[dst[e],h,:]); running segment-max.
__global__ void edge_scores(const float* __restrict__ Q, const float* __restrict__ K,
                            const int* __restrict__ src, const int* __restrict__ dst,
                            float* __restrict__ sc, unsigned* __restrict__ mbuf, int E) {
    long idx = (long)blockIdx.x * 256 + threadIdx.x;
    if (idx >= (long)E * HH) return;
    int e = (int)(idx >> 3), h = (int)(idx & 7);
    int s = src[e], d = dst[e];
    const float4* kp = (const float4*)(K + s * DD + h * DHH);
    const float4* qp = (const float4*)(Q + d * DD + h * DHH);
    float acc = 0.0f;
#pragma unroll
    for (int i = 0; i < 4; ++i) {
        float4 kv = kp[i], qv = qp[i];
        acc += kv.x * qv.x + kv.y * qv.y + kv.z * qv.z + kv.w * qv.w;
    }
    sc[idx] = acc;
    atomicMax(&mbuf[d * HH + h], enc_key(acc));
}

// ex = exp(score - m[dst]); denom[dst] += ex (overwrite score buffer with ex).
__global__ void edge_exp(float* __restrict__ sc, const unsigned* __restrict__ mbuf,
                         float* __restrict__ denom, const int* __restrict__ dst, int E) {
    long idx = (long)blockIdx.x * 256 + threadIdx.x;
    if (idx >= (long)E * HH) return;
    int e = (int)(idx >> 3), h = (int)(idx & 7);
    int d = dst[e];
    float ex = __expf(sc[idx] - dec_key(mbuf[d * HH + h]));
    sc[idx] = ex;
    unsafeAtomicAdd(&denom[d * HH + h], ex);
}

// agg[dst, c] += V[src, c] * p[e, c/16]   (one thread per edge-channel)
__global__ void edge_agg(const float* __restrict__ V, const float* __restrict__ sc,
                         const float* __restrict__ denom,
                         const int* __restrict__ src, const int* __restrict__ dst,
                         float* __restrict__ agg, int E) {
    long idx = (long)blockIdx.x * 256 + threadIdx.x;
    if (idx >= (long)E * DD) return;
    int e = (int)(idx >> 7), c = (int)(idx & 127);
    int s = src[e], d = dst[e];
    int h = c >> 4;
    float p = sc[e * HH + h] / denom[d * HH + h];
    unsafeAtomicAdd(&agg[d * DD + c], V[s * DD + c] * p);
}

// Row LayerNorm: one wave per row (wave32, 4 cols/lane), shfl_xor reductions.
__global__ __launch_bounds__(256)
void layernorm_k(const float* __restrict__ y, const float* __restrict__ gamma,
                 const float* __restrict__ beta, float* __restrict__ out, int n) {
    int lane = threadIdx.x & 31;
    int row  = blockIdx.x * 8 + (threadIdx.x >> 5);
    if (row >= n) return;
    const float* yr = y + row * DD;
    float v[4];
    float s = 0.0f;
#pragma unroll
    for (int i = 0; i < 4; ++i) { v[i] = yr[lane + 32 * i]; s += v[i]; }
#pragma unroll
    for (int o = 16; o > 0; o >>= 1) s += __shfl_xor(s, o, 32);
    float mu = s * (1.0f / 128.0f);
    float q = 0.0f;
#pragma unroll
    for (int i = 0; i < 4; ++i) { v[i] -= mu; q += v[i] * v[i]; }
#pragma unroll
    for (int o = 16; o > 0; o >>= 1) q += __shfl_xor(q, o, 32);
    float inv = rsqrtf(q * (1.0f / 128.0f) + 1e-12f);
#pragma unroll
    for (int i = 0; i < 4; ++i) {
        int c = lane + 32 * i;
        out[row * DD + c] = v[i] * inv * gamma[c] + beta[c];
    }
}

// ---------------------------------------------------------------------------
extern "C" void kernel_launch(void* const* d_in, const int* in_sizes, int n_in,
                              void* d_out, int out_size, void* d_ws, size_t ws_size,
                              hipStream_t stream) {
    const float* h     = (const float*)d_in[0];
    const float* Wq    = (const float*)d_in[1];
    const float* bq    = (const float*)d_in[2];
    const float* Wk    = (const float*)d_in[3];
    const float* bk    = (const float*)d_in[4];
    const float* Wv    = (const float*)d_in[5];
    const float* bv    = (const float*)d_in[6];
    const float* Wo    = (const float*)d_in[7];
    const float* bo    = (const float*)d_in[8];
    const float* gamma = (const float*)d_in[9];
    const float* beta  = (const float*)d_in[10];
    const int*   src   = (const int*)d_in[11];
    const int*   dst   = (const int*)d_in[12];

    const int n = in_sizes[0] / DD;
    const int E = in_sizes[11];

    // Workspace carve-up (floats)
    float*    ws    = (float*)d_ws;
    size_t    ND    = (size_t)n * DD;
    float*    Q     = ws;
    float*    K     = Q + ND;
    float*    V     = K + ND;
    float*    agg   = V + ND;
    float*    y     = agg + ND;
    float*    sc    = y + ND;                      // E*H
    float*    denom = sc + (size_t)E * HH;         // n*H
    unsigned* mbuf  = (unsigned*)(denom + (size_t)n * HH);  // n*H

    const int nTile = (n + 15) / 16;

    {   // zero/identity-init scratch (workspace is poisoned by harness)
        long total  = (long)n * DD;
        int  blocks = (int)((total + 255) / 256);
        init_scratch<<<blocks, 256, 0, stream>>>(agg, denom, mbuf, n);
    }

    gemm_bf16_wmma<false><<<nTile, 256, 0, stream>>>(h, Wq, bq, nullptr, Q, n);
    gemm_bf16_wmma<false><<<nTile, 256, 0, stream>>>(h, Wk, bk, nullptr, K, n);
    gemm_bf16_wmma<false><<<nTile, 256, 0, stream>>>(h, Wv, bv, nullptr, V, n);

    {
        long total  = (long)E * HH;
        int  blocks = (int)((total + 255) / 256);
        edge_scores<<<blocks, 256, 0, stream>>>(Q, K, src, dst, sc, mbuf, E);
        edge_exp   <<<blocks, 256, 0, stream>>>(sc, mbuf, denom, dst, E);
    }
    {
        long total  = (long)E * DD;
        int  blocks = (int)((total + 255) / 256);
        edge_agg<<<blocks, 256, 0, stream>>>(V, sc, denom, src, dst, agg, E);
    }

    gemm_bf16_wmma<true><<<nTile, 256, 0, stream>>>(agg, Wo, bo, h, y, n);
    layernorm_k<<<(n + 7) / 8, 256, 0, stream>>>(y, gamma, beta, (float*)d_out, n);
}
